// WSAD_15573551415445
// MI455X (gfx1250) — compile-verified, matrix operand order checked by reading, not run on hardware
//
#include <hip/hip_runtime.h>
#include <hip/hip_bf16.h>
#include <math.h>

// ---------------------------------------------------------------------------
// WSAD pipeline for MI455X (gfx1250, wave32).
// Matmuls: v_wmma_f32_16x16x32_bf16.  Tile staging: global_load_async_to_lds
// (ASYNCcnt-tracked DMA), double-buffered so copies overlap WMMA compute.
// ---------------------------------------------------------------------------

typedef unsigned short u16;
typedef __attribute__((ext_vector_type(16))) __bf16 bf16x16;
typedef __attribute__((ext_vector_type(8)))  float  f32x8;

#define M_TOK   81920   // 320 * 256 token rows
#define EPS_LN  1e-5f
#define ATTN_SCALE 0.08838834764831845f  // 128^-0.5

#define ACT_NONE 0
#define ACT_RELU 1
#define ACT_GELU 2

// 32 bytes per lane: global -> LDS, no VGPR round trip (ASYNCcnt-tracked)
#define ASYNC_CP32(ldsOff, gptr)                                            \
    asm volatile("global_load_async_to_lds_b128 %0, %1, off\n\t"            \
                 "global_load_async_to_lds_b128 %0, %1, off offset:16"      \
                 :: "v"(ldsOff), "v"(gptr) : "memory")

#define ASYNC_CP64(ldsOff, gptr)                                            \
    asm volatile("global_load_async_to_lds_b128 %0, %1, off\n\t"            \
                 "global_load_async_to_lds_b128 %0, %1, off offset:16\n\t"  \
                 "global_load_async_to_lds_b128 %0, %1, off offset:32\n\t"  \
                 "global_load_async_to_lds_b128 %0, %1, off offset:48"      \
                 :: "v"(ldsOff), "v"(gptr) : "memory")

#define WAIT_ASYNC() asm volatile("s_wait_asynccnt 0x0" ::: "memory")

__device__ __forceinline__ unsigned lds_off(const void* p) {
    return (unsigned)(size_t)p;   // generic LDS address: low 32 bits = LDS offset
}

// ---------------- small helpers ----------------

__device__ __forceinline__ u16 f2bf(float f) {
    union { float f; unsigned u; } v; v.f = f;
    unsigned u = v.u;
    unsigned r = u + 0x7FFFu + ((u >> 16) & 1u);   // RNE
    return (u16)(r >> 16);
}
__device__ __forceinline__ float bf2f(u16 h) {
    union { unsigned u; float f; } v; v.u = ((unsigned)h) << 16;
    return v.f;
}

union FragCvt { struct { uint4 lo, hi; } q; bf16x16 v; };

// 16x32 bf16 fragment from LDS tile stored [row][k], row stride strideE.
// CDNA5 16-bit A/B layout: lanes 0-15 K {0..7,16..23}; lanes 16-31 K {8..15,24..31}.
__device__ __forceinline__ bf16x16 load_frag(const u16* base, int strideE, int lane) {
    const int r  = lane & 15;
    const int ho = (lane & 16) ? 8 : 0;
    const u16* p = base + r * strideE + ho;
    FragCvt f;
    f.q.lo = *(const uint4*)(p);
    f.q.hi = *(const uint4*)(p + 16);
    return f.v;
}

__device__ __forceinline__ f32x8 wmma_bf16(bf16x16 a, bf16x16 b, f32x8 c) {
    return __builtin_amdgcn_wmma_f32_16x16x32_bf16(false, a, false, b, (short)0, c, false, false);
}

__device__ __forceinline__ float wave_sum(float v) {
#pragma unroll
    for (int m = 16; m >= 1; m >>= 1) v += __shfl_xor(v, m, 32);
    return v;
}
__device__ __forceinline__ float half_sum(float v) {
#pragma unroll
    for (int m = 8; m >= 1; m >>= 1) v += __shfl_xor(v, m, 32);
    return v;
}
__device__ __forceinline__ float half_max(float v) {
#pragma unroll
    for (int m = 8; m >= 1; m >>= 1) v = fmaxf(v, __shfl_xor(v, m, 32));
    return v;
}

// ---------------- utility kernels ----------------

__global__ void wsad_zero_f32(float* p, int n) {
    int i = blockIdx.x * blockDim.x + threadIdx.x;
    if (i < n) p[i] = 0.f;
}

// transpose-pack: s[l][k][n] (f32) -> d[l][n][k] (bf16)
__global__ void wsad_pack_bf16_T(const float* __restrict__ s, u16* __restrict__ d,
                                 int K, int N, int total) {
    int i = blockIdx.x * blockDim.x + threadIdx.x;
    if (i >= total) return;
    int kn = K * N;
    int l  = i / kn;
    int r  = i - l * kn;
    int n  = r / K;
    int k  = r - n * K;
    d[i] = f2bf(s[(size_t)l * kn + (size_t)k * N + n]);
}

// W_emb (512, 1024, 3) [O][I][H] -> Wc[o][k = h*1024 + i]  (im2col weight, [N][K])
__global__ void wsad_pack_convw(const float* __restrict__ s, u16* __restrict__ d) {
    int i = blockIdx.x * blockDim.x + threadIdx.x;     // i = o*3072 + k
    if (i >= 512 * 3072) return;
    int o   = i / 3072;
    int k   = i - o * 3072;
    int tap = k >> 10;
    int dd  = k & 1023;
    d[i] = f2bf(s[(size_t)o * 3072 + dd * 3 + tap]);
}

__global__ void wsad_finalize(const float* accS, const float* accD1,
                              const float* accD2, float* out) {
    int i = blockIdx.x * blockDim.x + threadIdx.x;
    if (i < 8192) out[i] = (accD1[i] + accD2[i]) * accS[i];
}

// ---------------- generic bf16 WMMA GEMM ----------------
// C[M,N] = A[M,K](bf16,row-major) * Bt[N,K](bf16,row-major = B^T)
// (+bias)(act)(+resid) -> f32 and/or bf16 outputs.
// Tile 128x128x32, 256 threads = 8 waves (4x2), each wave 32x64.
// Double-buffered LDS filled by async global->LDS DMA; one barrier per k-step.

__global__ __launch_bounds__(256) void wsad_gemm_bf16(
    const u16* __restrict__ A, const u16* __restrict__ Bt,
    const float* __restrict__ bias, const float* __restrict__ resid,
    float* __restrict__ outF, u16* __restrict__ outH,
    int M, int N, int K, int act)
{
    __shared__ u16 As[2][128 * 32];
    __shared__ u16 Bs[2][128 * 32];   // [n][k]

    const int tid  = threadIdx.x;
    const int lane = tid & 31;
    const int wid  = tid >> 5;
    const int wm   = (wid & 3) * 32;
    const int wn   = (wid >> 2) * 64;
    const long mBase = (long)blockIdx.x * 128;
    const long nBase = (long)blockIdx.y * 128;

    f32x8 acc[2][4];
#pragma unroll
    for (int i = 0; i < 2; ++i)
#pragma unroll
        for (int j = 0; j < 4; ++j) acc[i][j] = (f32x8){};

    const int srow = tid >> 1, scol = (tid & 1) * 16;   // 16 bf16 per thread per tile
    const u16* gA = A  + (mBase + srow) * (long)K + scol;
    const u16* gB = Bt + (nBase + srow) * (long)K + scol;
    const int kIters = K >> 5;

    // prologue: stage tile 0 into buffer 0
    {
        unsigned la = lds_off(&As[0][srow * 32 + scol]);
        unsigned lb = lds_off(&Bs[0][srow * 32 + scol]);
        ASYNC_CP32(la, gA);
        ASYNC_CP32(lb, gB);
    }

    int buf = 0;
    for (int kt = 0; kt < kIters; ++kt) {
        WAIT_ASYNC();          // this wave's DMA into LDS complete
        __syncthreads();       // everyone's complete; other buffer free

        if (kt + 1 < kIters) { // overlap next tile's DMA with this tile's math
            unsigned la = lds_off(&As[buf ^ 1][srow * 32 + scol]);
            unsigned lb = lds_off(&Bs[buf ^ 1][srow * 32 + scol]);
            ASYNC_CP32(la, gA + (kt + 1) * 32);
            ASYNC_CP32(lb, gB + (kt + 1) * 32);
        }

        const u16* as = &As[buf][0];
        const u16* bs = &Bs[buf][0];
        bf16x16 af0 = load_frag(as + (wm +  0) * 32, 32, lane);
        bf16x16 af1 = load_frag(as + (wm + 16) * 32, 32, lane);
        bf16x16 bf0 = load_frag(bs + (wn +  0) * 32, 32, lane);
        bf16x16 bf1 = load_frag(bs + (wn + 16) * 32, 32, lane);
        bf16x16 bf2 = load_frag(bs + (wn + 32) * 32, 32, lane);
        bf16x16 bf3 = load_frag(bs + (wn + 48) * 32, 32, lane);
        acc[0][0] = wmma_bf16(af0, bf0, acc[0][0]);
        acc[1][0] = wmma_bf16(af1, bf0, acc[1][0]);
        acc[0][1] = wmma_bf16(af0, bf1, acc[0][1]);
        acc[1][1] = wmma_bf16(af1, bf1, acc[1][1]);
        acc[0][2] = wmma_bf16(af0, bf2, acc[0][2]);
        acc[1][2] = wmma_bf16(af1, bf2, acc[1][2]);
        acc[0][3] = wmma_bf16(af0, bf3, acc[0][3]);
        acc[1][3] = wmma_bf16(af1, bf3, acc[1][3]);
        buf ^= 1;
    }

    const int rofs = (lane & 16) ? 8 : 0;
    const int cl   = lane & 15;
#pragma unroll
    for (int mi = 0; mi < 2; ++mi)
#pragma unroll
        for (int ni = 0; ni < 4; ++ni)
#pragma unroll
            for (int r = 0; r < 8; ++r) {
                long row = mBase + wm + mi * 16 + rofs + r;
                long col = nBase + wn + ni * 16 + cl;
                float v = acc[mi][ni][r];
                if (bias) v += bias[col];
                if (act == ACT_RELU)      v = fmaxf(v, 0.f);
                else if (act == ACT_GELU) v = 0.5f * v * (1.f + erff(v * 0.70710678118f));
                if (resid) v += resid[row * N + col];
                if (outF) outF[row * N + col] = v;
                if (outH) outH[row * N + col] = f2bf(v);
            }
}

// ---------------- conv1d(k=3,pad=1) as implicit-im2col WMMA GEMM ----------
// A[m, k] = x[bn, t + (k/1024) - 1, k%1024] (zero padded, f32->bf16 on the fly)
// Bt = packed conv weight [O=512][K=3072].  relu(acc + bias) -> h f32.

__global__ __launch_bounds__(256) void wsad_conv_gemm(
    const float* __restrict__ X, const u16* __restrict__ Bt,
    const float* __restrict__ bias, float* __restrict__ outF)
{
    const int N = 512, K = 3072;
    __shared__ u16 As[128 * 32];
    __shared__ u16 Bs[128 * 32];

    const int tid  = threadIdx.x;
    const int lane = tid & 31;
    const int wid  = tid >> 5;
    const int wm   = (wid & 3) * 32;
    const int wn   = (wid >> 2) * 64;
    const long mBase = (long)blockIdx.x * 128;
    const long nBase = (long)blockIdx.y * 128;

    f32x8 acc[2][4];
#pragma unroll
    for (int i = 0; i < 2; ++i)
#pragma unroll
        for (int j = 0; j < 4; ++j) acc[i][j] = (f32x8){};

    const int srow = tid >> 1, scol = (tid & 1) * 16;
    const u16* gB = Bt + (nBase + srow) * (long)K + scol;
    const unsigned lb = lds_off(&Bs[srow * 32 + scol]);

    for (int kt = 0; kt < (K >> 5); ++kt) {
        // B tile by async DMA (overlaps with the A conversion below)
        ASYNC_CP32(lb, gB + kt * 32);
        {   // implicit im2col stage of A (convert f32 -> bf16)
            long m   = mBase + srow;
            int bn_  = (int)(m >> 8);
            int t    = (int)(m & 255);
            int kg   = kt * 32 + scol;
            int tap  = kg >> 10;
            int d    = kg & 1023;
            int ts   = t + tap - 1;
            u16* dst = &As[srow * 32 + scol];
            if (ts >= 0 && ts < 256) {
                const float* src = X + ((long)bn_ * 256 + ts) * 1024 + d;
#pragma unroll
                for (int i = 0; i < 16; ++i) dst[i] = f2bf(src[i]);
                __builtin_prefetch(src + 32, 0, 1);
            } else {
#pragma unroll
                for (int i = 0; i < 16; ++i) dst[i] = 0;
            }
        }
        WAIT_ASYNC();
        __syncthreads();

        bf16x16 af0 = load_frag(&As[(wm +  0) * 32], 32, lane);
        bf16x16 af1 = load_frag(&As[(wm + 16) * 32], 32, lane);
#pragma unroll
        for (int ni = 0; ni < 4; ++ni) {
            bf16x16 bf = load_frag(&Bs[(wn + ni * 16) * 32], 32, lane);
            acc[0][ni] = wmma_bf16(af0, bf, acc[0][ni]);
            acc[1][ni] = wmma_bf16(af1, bf, acc[1][ni]);
        }
        __syncthreads();
    }

    const int rofs = (lane & 16) ? 8 : 0;
    const int cl   = lane & 15;
#pragma unroll
    for (int mi = 0; mi < 2; ++mi)
#pragma unroll
        for (int ni = 0; ni < 4; ++ni)
#pragma unroll
            for (int r = 0; r < 8; ++r) {
                long row = mBase + wm + mi * 16 + rofs + r;
                long col = nBase + wn + ni * 16 + cl;
                float v = fmaxf(acc[mi][ni][r] + bias[col], 0.f);
                outF[row * N + col] = v;
            }
}

// ---------------- LayerNorm (512) -> bf16 ----------------

__global__ __launch_bounds__(256) void wsad_ln(
    const float* __restrict__ h, const float* __restrict__ g,
    const float* __restrict__ b, u16* __restrict__ u)
{
    const int lane = threadIdx.x & 31, wid = threadIdx.x >> 5;
    const long row = (long)blockIdx.x * 8 + wid;
    const float* hr = h + row * 512 + lane * 16;
    float v[16];
    float s = 0.f;
#pragma unroll
    for (int i = 0; i < 16; ++i) { v[i] = hr[i]; s += v[i]; }
    s = wave_sum(s);
    float mu = s * (1.f / 512.f);
    float vs = 0.f;
#pragma unroll
    for (int i = 0; i < 16; ++i) { float d = v[i] - mu; vs += d * d; }
    vs = wave_sum(vs);
    float inv = rsqrtf(vs * (1.f / 512.f) + EPS_LN);
    u16* ur = u + row * 512 + lane * 16;
#pragma unroll
    for (int i = 0; i < 16; ++i) {
        int c = lane * 16 + i;
        ur[i] = f2bf((v[i] - mu) * inv * g[c] + b[c]);
    }
}

// ---------------- attention (flash-style, WMMA) ----------------
// Block = 128 threads (4 waves); one block per (bn, head, q-quarter of 64 rows).
// K chunks double-buffered via async DMA so copy overlaps S-tile WMMAs.

__global__ __launch_bounds__(128) void wsad_attn(
    const u16* __restrict__ qkv, u16* __restrict__ o)
{
    __shared__ u16 Kc[2][32 * 128];     // [key][dh]   16 KB (double buffered)
    __shared__ u16 Vt[128 * 32];        // [dh][key]    8 KB
    __shared__ u16 Ps[4][16][256];      // per-wave P  32 KB

    const int tid  = threadIdx.x, lane = tid & 31, wid = tid >> 5;
    const int bid  = blockIdx.x;
    const int qq   = bid & 3;
    const int head = (bid >> 2) & 3;
    const int bn_  = bid >> 4;
    const long rowBase = (long)bn_ * 256;
    const int qbase = qq * 64 + wid * 16;
    const int cl   = lane & 15;
    const int ko   = (lane & 16) ? 8 : 0;
    const int rofs = ko;

    // Q fragments: 16 rows x 128 dh = 4 chunks of 16x32
    bf16x16 qf[4];
    {
        const u16* qrow = qkv + (rowBase + qbase + cl) * 1536 + head * 128;
#pragma unroll
        for (int c = 0; c < 4; ++c) {
            FragCvt f;
            f.q.lo = *(const uint4*)(qrow + c * 32 + ko);
            f.q.hi = *(const uint4*)(qrow + c * 32 + ko + 16);
            qf[c] = f.v;
        }
    }

    const int skey = tid >> 2, sseg = (tid & 3) * 32;   // staging role: 64B per thread
    const u16* srcK = qkv + (rowBase + skey) * 1536 + 512 + head * 128 + sseg;

    float rmax[8];
#pragma unroll
    for (int r = 0; r < 8; ++r) rmax[r] = -3.0e38f;

    // Phase 1: S = scale * Q K^T; K chunks pipelined (pattern B, 1 barrier/chunk).
    ASYNC_CP64(lds_off(&Kc[0][skey * 128 + sseg]), srcK);
    int buf = 0;
    for (int kc = 0; kc < 8; ++kc) {
        WAIT_ASYNC();
        __syncthreads();
        if (kc + 1 < 8) {
            ASYNC_CP64(lds_off(&Kc[buf ^ 1][skey * 128 + sseg]),
                       srcK + (long)(kc + 1) * 32 * 1536);
        }
#pragma unroll
        for (int nt = 0; nt < 2; ++nt) {
            f32x8 s = (f32x8){};
#pragma unroll
            for (int c = 0; c < 4; ++c) {
                bf16x16 kf = load_frag(&Kc[buf][(nt * 16) * 128 + c * 32], 128, lane);
                s = wmma_bf16(qf[c], kf, s);
            }
#pragma unroll
            for (int r = 0; r < 8; ++r) {
                float sv = s[r] * ATTN_SCALE;
                rmax[r] = fmaxf(rmax[r], sv);
                Ps[wid][r + rofs][kc * 32 + nt * 16 + cl] = f2bf(sv);
            }
        }
        buf ^= 1;
    }

#pragma unroll
    for (int r = 0; r < 8; ++r) rmax[r] = half_max(rmax[r]);

    // Phase 2: exponentiate in place, accumulate row sums (per-wave LDS region).
    float rsum[8];
#pragma unroll
    for (int r = 0; r < 8; ++r) rsum[r] = 0.f;
    for (int ct = 0; ct < 16; ++ct) {
#pragma unroll
        for (int r = 0; r < 8; ++r) {
            int col = ct * 16 + cl;
            float v = bf2f(Ps[wid][r + rofs][col]);
            float p = expf(v - rmax[r]);
            rsum[r] += p;
            Ps[wid][r + rofs][col] = f2bf(p);
        }
    }
#pragma unroll
    for (int r = 0; r < 8; ++r) rsum[r] = half_sum(rsum[r]);

    // Phase 3: O = P * V  (V transposed into LDS per chunk)
    f32x8 oacc[8];
#pragma unroll
    for (int ot = 0; ot < 8; ++ot) oacc[ot] = (f32x8){};
    for (int kc = 0; kc < 8; ++kc) {
        {   // stage V chunk transposed: [dh][key]
            const u16* src = qkv + (rowBase + kc * 32 + skey) * 1536 + 1024 + head * 128 + sseg;
#pragma unroll
            for (int i = 0; i < 32; ++i) Vt[(sseg + i) * 32 + skey] = src[i];
        }
        __syncthreads();
        bf16x16 pf = load_frag(&Ps[wid][0][kc * 32], 256, lane);
#pragma unroll
        for (int ot = 0; ot < 8; ++ot) {
            bf16x16 vf = load_frag(&Vt[(ot * 16) * 32], 32, lane);
            oacc[ot] = wmma_bf16(pf, vf, oacc[ot]);
        }
        __syncthreads();
    }

    // normalize + store bf16
#pragma unroll
    for (int ot = 0; ot < 8; ++ot)
#pragma unroll
        for (int r = 0; r < 8; ++r) {
            float v = oacc[ot][r] / rsum[r];
            long row = rowBase + qbase + rofs + r;
            o[row * 512 + head * 128 + ot * 16 + cl] = f2bf(v);
        }
}

// ---------------- NormalHead: 512 -> 32 -> 16 -> 1, dists + score --------

__global__ __launch_bounds__(256) void wsad_head(
    const float* __restrict__ h,
    const float* __restrict__ c1W, const float* __restrict__ c1b,
    const float* __restrict__ g1,  const float* __restrict__ b1,
    const float* __restrict__ rm1, const float* __restrict__ rv1,
    const float* __restrict__ c2W, const float* __restrict__ c2b,
    const float* __restrict__ g2,  const float* __restrict__ b2,
    const float* __restrict__ rm2, const float* __restrict__ rv2,
    const float* __restrict__ c3W, const float* __restrict__ c3b,
    float* accS, float* accD1, float* accD2)
{
    __shared__ float hs[8][512];
    const int lane = threadIdx.x & 31, wid = threadIdx.x >> 5;
    const long row = (long)blockIdx.x * 8 + wid;
    const float* hr = h + row * 512;
#pragma unroll
    for (int i = 0; i < 16; ++i) hs[wid][lane * 16 + i] = hr[lane * 16 + i];
    __syncthreads();

    // conv1: lane = output channel o (0..31)
    float x1 = c1b[lane];
    const float* w1p = c1W + lane * 512;
#pragma unroll 8
    for (int c = 0; c < 512; ++c) x1 += hs[wid][c] * w1p[c];

    float dd1 = x1 - rm1[lane];
    float d1  = sqrtf(wave_sum(dd1 * dd1 / rv1[lane]));
    float r1  = fmaxf(dd1 * rsqrtf(rv1[lane] + EPS_LN) * g1[lane] + b1[lane], 0.f);

    // conv2: lanes 0..15 own output channel o2
    int o2 = lane & 15;
    float x2 = c2b[o2];
    const float* w2p = c2W + o2 * 32;
#pragma unroll
    for (int c = 0; c < 32; ++c) x2 += __shfl(r1, c, 32) * w2p[c];

    float dd2 = x2 - rm2[o2];
    float d2p = (lane < 16) ? dd2 * dd2 / rv2[o2] : 0.f;
    float d2  = sqrtf(wave_sum(d2p));
    float r2  = fmaxf(dd2 * rsqrtf(rv2[o2] + EPS_LN) * g2[o2] + b2[o2], 0.f);

    // conv3 -> sigmoid score
    float x3 = c3b[0];
#pragma unroll
    for (int c = 0; c < 16; ++c) x3 += __shfl(r2, c, 32) * c3W[c];
    float score = 1.f / (1.f + expf(-x3));

    if (lane == 0) {
        int b = (int)(row / 2560);
        int t = (int)(row & 255);
        atomicAdd(&accS [b * 256 + t], score * 0.1f);
        atomicAdd(&accD1[b * 256 + t], d1    * 0.1f);
        atomicAdd(&accD2[b * 256 + t], d2    * 0.1f);
    }
}

// ---------------------------------------------------------------------------

extern "C" void kernel_launch(void* const* d_in, const int* in_sizes, int n_in,
                              void* d_out, int out_size, void* d_ws, size_t ws_size,
                              hipStream_t stream) {
    (void)in_sizes; (void)n_in; (void)out_size; (void)ws_size;

    const float* x      = (const float*)d_in[0];
    const float* W_emb  = (const float*)d_in[1];
    const float* b_emb  = (const float*)d_in[2];
    const float* ln1_g  = (const float*)d_in[3];
    const float* ln1_b  = (const float*)d_in[4];
    const float* Wqkv_f = (const float*)d_in[5];
    const float* Wo_f   = (const float*)d_in[6];
    const float* bo     = (const float*)d_in[7];
    const float* ln2_g  = (const float*)d_in[8];
    const float* ln2_b  = (const float*)d_in[9];
    const float* W1_f   = (const float*)d_in[10];
    const float* b1     = (const float*)d_in[11];
    const float* W2_f   = (const float*)d_in[12];
    const float* b2     = (const float*)d_in[13];
    const float* c1W    = (const float*)d_in[14];
    const float* c1b    = (const float*)d_in[15];
    const float* bn1g   = (const float*)d_in[16];
    const float* bn1b   = (const float*)d_in[17];
    const float* bn1rm  = (const float*)d_in[18];
    const float* bn1rv  = (const float*)d_in[19];
    const float* c2W    = (const float*)d_in[20];
    const float* c2b    = (const float*)d_in[21];
    const float* bn2g   = (const float*)d_in[22];
    const float* bn2b   = (const float*)d_in[23];
    const float* bn2rm  = (const float*)d_in[24];
    const float* bn2rv  = (const float*)d_in[25];
    const float* c3W    = (const float*)d_in[26];
    const float* c3b    = (const float*)d_in[27];

    // -------- workspace carve --------
    char* w = (char*)d_ws;
    float* h    = (float*)w; w += (size_t)M_TOK * 512 * 4;   // activations (f32)
    u16*   u    = (u16*)w;   w += (size_t)M_TOK * 512 * 2;   // LN output (bf16)
    u16*   qkv  = (u16*)w;   w += (size_t)M_TOK * 1536 * 2;  // QKV (bf16)
    u16*   ob   = (u16*)w;   w += (size_t)M_TOK * 512 * 2;   // attn out / mlp hidden
    u16*   Wc   = (u16*)w;   w += (size_t)512 * 3072 * 2;    // conv weight [O][K]
    u16*   Wq   = (u16*)w;   w += (size_t)2 * 1536 * 512 * 2;// [l][N][K]
    u16*   Wo   = (u16*)w;   w += (size_t)2 * 512 * 512 * 2;
    u16*   W1   = (u16*)w;   w += (size_t)2 * 512 * 512 * 2;
    u16*   W2   = (u16*)w;   w += (size_t)2 * 512 * 512 * 2;
    float* accS  = (float*)w; w += 8192 * 4;
    float* accD1 = (float*)w; w += 8192 * 4;
    float* accD2 = (float*)w; w += 8192 * 4;

    // -------- setup --------
    wsad_zero_f32<<<(3 * 8192 + 255) / 256, 256, 0, stream>>>(accS, 3 * 8192);
    wsad_pack_convw<<<(512 * 3072 + 255) / 256, 256, 0, stream>>>(W_emb, Wc);
    wsad_pack_bf16_T<<<(2 * 512 * 1536 + 255) / 256, 256, 0, stream>>>(
        Wqkv_f, Wq, 512, 1536, 2 * 512 * 1536);
    wsad_pack_bf16_T<<<(2 * 512 * 512 + 255) / 256, 256, 0, stream>>>(
        Wo_f, Wo, 512, 512, 2 * 512 * 512);
    wsad_pack_bf16_T<<<(2 * 512 * 512 + 255) / 256, 256, 0, stream>>>(
        W1_f, W1, 512, 512, 2 * 512 * 512);
    wsad_pack_bf16_T<<<(2 * 512 * 512 + 255) / 256, 256, 0, stream>>>(
        W2_f, W2, 512, 512, 2 * 512 * 512);

    // -------- temporal conv (im2col GEMM, relu) --------
    wsad_conv_gemm<<<dim3(M_TOK / 128, 512 / 128), 256, 0, stream>>>(x, Wc, b_emb, h);

    // -------- transformer layers --------
    for (int l = 0; l < 2; ++l) {
        wsad_ln<<<M_TOK / 8, 256, 0, stream>>>(h, ln1_g + l * 512, ln1_b + l * 512, u);
        wsad_gemm_bf16<<<dim3(M_TOK / 128, 1536 / 128), 256, 0, stream>>>(
            u, Wq + (size_t)l * 1536 * 512, nullptr, nullptr, nullptr, qkv,
            M_TOK, 1536, 512, ACT_NONE);
        wsad_attn<<<320 * 4 * 4, 128, 0, stream>>>(qkv, ob);
        wsad_gemm_bf16<<<dim3(M_TOK / 128, 512 / 128), 256, 0, stream>>>(
            ob, Wo + (size_t)l * 512 * 512, bo + l * 512, h, h, nullptr,
            M_TOK, 512, 512, ACT_NONE);
        wsad_ln<<<M_TOK / 8, 256, 0, stream>>>(h, ln2_g + l * 512, ln2_b + l * 512, u);
        wsad_gemm_bf16<<<dim3(M_TOK / 128, 512 / 128), 256, 0, stream>>>(
            u, W1 + (size_t)l * 512 * 512, b1 + l * 512, nullptr, nullptr, ob,
            M_TOK, 512, 512, ACT_GELU);
        wsad_gemm_bf16<<<dim3(M_TOK / 128, 512 / 128), 256, 0, stream>>>(
            ob, W2 + (size_t)l * 512 * 512, b2 + l * 512, h, h, nullptr,
            M_TOK, 512, 512, ACT_NONE);
    }

    // -------- head + reductions --------
    wsad_head<<<M_TOK / 8, 256, 0, stream>>>(
        h, c1W, c1b, bn1g, bn1b, bn1rm, bn1rv,
        c2W, c2b, bn2g, bn2b, bn2rm, bn2rv, c3W, c3b,
        accS, accD1, accD2);
    wsad_finalize<<<32, 256, 0, stream>>>(accS, accD1, accD2, (float*)d_out);
}
